// BPCAPooling_283467842314
// MI455X (gfx1250) — matrix operand order repaired and checked by reading, not compiled, not run on hardware
//
#include <hip/hip_runtime.h>
#include <hip/hip_bf16.h>

typedef __attribute__((ext_vector_type(16))) __bf16 v16bf;
typedef __attribute__((ext_vector_type(8)))  float  v8f;

#define POOLK       4               // POOL*POOL matrix columns
#define MROWS       802816          // rows per batch = 112*112*64
#define NBATCH      8
#define NWG         196             // pass-1 workgroups per batch (802816/196 = 4096)
#define ROWS_PER_WG (MROWS / NWG)   // 4096
#define P1_ITERS    (ROWS_PER_WG / 256)  // 16
#define NWAVES      8
#define TROWS       16              // tile rows: 0..3 live columns, 4..15 pinned zero
#define PART_STRIDE 20              // 16 Gram + 4 column sums
#define PART_FLOATS (NBATCH * NWG * PART_STRIDE)
// finalize region: per batch 8 floats at PART_FLOATS + b*8 : w0..w3, bias

// One 32-row step: load row quad, accumulate column sums, stage bf16 transpose
// tile, gather WMMA fragments (all lanes; rows >=4 are permanently zero), MAC.
static __device__ __forceinline__ void p1_step(
    const float4* __restrict__ in4, int row,
    __bf16 (*tile)[32], int lane, int mlane, int half,
    float4& s, v8f& cacc) {
    const float4 a = in4[row];
    s.x += a.x; s.y += a.y; s.z += a.z; s.w += a.w;

    tile[0][lane] = (__bf16)a.x;
    tile[1][lane] = (__bf16)a.y;
    tile[2][lane] = (__bf16)a.z;
    tile[3][lane] = (__bf16)a.w;
    __builtin_amdgcn_wave_barrier();

    v16bf af, bf;
    // A (16x32, M=column index k, K=data row):
    // lanes 0-15: K = 0..15 ; lanes 16-31: K = 16..31
#pragma unroll
    for (int e = 0; e < 8; ++e) {
        af[e]     = tile[mlane][half * 16 + e];
        af[8 + e] = tile[mlane][half * 16 + 8 + e];
    }
    // B (32x16, N=column index l): lane n<16: K=0..15 ; lane n+16: K=16..31
#pragma unroll
    for (int e = 0; e < 16; ++e)
        bf[e] = tile[mlane][half * 16 + e];
    __builtin_amdgcn_wave_barrier();

    // D[k][l] accumulates C_kl = sum over rows of a_k * a_l  (f32 accumulate)
    cacc = __builtin_amdgcn_wmma_f32_16x16x32_bf16(
               false, af, false, bf, (short)0, cacc, false, false);
    __builtin_amdgcn_wave_barrier();
}

// ---------------------------------------------------------------------------
// Pass 1: raw moments. S_k = sum a_k (VALU+shfl reduce); C_kl = sum a_k a_l
// via v_wmma_f32_16x16x32_bf16 on per-wave LDS-transposed tiles, two
// independent accumulator chains to break the WMMA RAW dependency.
// ---------------------------------------------------------------------------
extern "C" __global__ __launch_bounds__(256)
void bpca_pass1(const float4* __restrict__ in4, float* __restrict__ part) {
    const int tid   = threadIdx.x;
    const int lane  = tid & 31;
    const int wid   = tid >> 5;
    const int b     = blockIdx.y;
    const int wg    = blockIdx.x;
    const int mlane = lane & 15;
    const int half  = lane >> 4;

    __shared__ __bf16 tile[NWAVES][TROWS][32];  // rows 4..15 stay zero forever
    __shared__ float  gred[NWAVES][4][4];
    __shared__ float  swav[NWAVES][4];

    // zero the pad rows once (zero everything: rows 0..3 are rewritten anyway)
    for (int i = tid; i < NWAVES * TROWS * 32; i += 256)
        ((__bf16*)tile)[i] = (__bf16)0.f;
    __syncthreads();

    float4 s = make_float4(0.f, 0.f, 0.f, 0.f);
    v8f cacc0, cacc1;
#pragma unroll
    for (int i = 0; i < 8; ++i) { cacc0[i] = 0.f; cacc1[i] = 0.f; }

    const int base = b * MROWS + wg * ROWS_PER_WG + tid;

    for (int it = 0; it < P1_ITERS; it += 2) {
        p1_step(in4, base + it * 256,       tile[wid], lane, mlane, half, s, cacc0);
        p1_step(in4, base + (it + 1) * 256, tile[wid], lane, mlane, half, s, cacc1);
    }
#pragma unroll
    for (int i = 0; i < 8; ++i) cacc0[i] += cacc1[i];

    // reduce column sums within wave
#pragma unroll
    for (int off = 16; off > 0; off >>= 1) {
        s.x += __shfl_down(s.x, off, 32);
        s.y += __shfl_down(s.y, off, 32);
        s.z += __shfl_down(s.z, off, 32);
        s.w += __shfl_down(s.w, off, 32);
    }
    if (lane == 0) {
        swav[wid][0] = s.x; swav[wid][1] = s.y;
        swav[wid][2] = s.z; swav[wid][3] = s.w;
    }
    // C/D layout: lane l (<16) holds N=l, VGPR k holds M=k -> G[k][l] = cacc0[k] @ lane l
    if (lane < POOLK) {
#pragma unroll
        for (int k = 0; k < POOLK; ++k) gred[wid][k][lane] = cacc0[k];
    }
    __syncthreads();

    float* pw = part + (b * NWG + wg) * PART_STRIDE;
    if (tid < 16) {
        float acc = 0.f;
        for (int w = 0; w < NWAVES; ++w) acc += gred[w][tid >> 2][tid & 3];
        pw[tid] = acc;
    } else if (tid < 20) {
        float acc = 0.f;
        for (int w = 0; w < NWAVES; ++w) acc += swav[w][tid - 16];
        pw[tid] = acc;
    }
}

// ---------------------------------------------------------------------------
// Finalize: reduce partials, build normalized Gram, shifted power iteration
// for the top eigenvector, emit projection weights w_k = v_k/sigma_k + bias.
// ---------------------------------------------------------------------------
extern "C" __global__ __launch_bounds__(32)
void bpca_finalize(float* __restrict__ ws) {
    const int b = blockIdx.x;
    const int t = threadIdx.x;
    __shared__ float red[PART_STRIDE];
    if (t < PART_STRIDE) {
        float acc = 0.f;
        const float* p = ws + b * NWG * PART_STRIDE + t;
        for (int w = 0; w < NWG; ++w) acc += p[w * PART_STRIDE];
        red[t] = acc;
    }
    __syncthreads();
    if (t == 0) {
        const float Mf = (float)MROWS;
        float m[4], isg[4], G[4][4];
        for (int k = 0; k < 4; ++k) m[k] = red[16 + k] / Mf;
        for (int k = 0; k < 4; ++k) {
            float var = red[k * 4 + k] / Mf - m[k] * m[k];
            isg[k] = (var > 0.f) ? rsqrtf(var) : 0.f;   // sigma==0 -> normed = 0
        }
        for (int k = 0; k < 4; ++k)
            for (int l = 0; l < 4; ++l)
                G[k][l] = (red[k * 4 + l] - Mf * m[k] * m[l]) * isg[k] * isg[l];

        // Gershgorin shift: G ~ M*I + O(sqrt(M)); shift so power iteration sees O(1) gaps
        float c = 0.f, dmin = G[0][0];
        for (int k = 0; k < 4; ++k) {
            float rs = 0.f;
            for (int l = 0; l < 4; ++l) if (l != k) rs += fabsf(G[k][l]);
            c = fmaxf(c, rs);
            dmin = fminf(dmin, G[k][k]);
        }
        const float sft = dmin - c;
        for (int k = 0; k < 4; ++k) G[k][k] -= sft;

        float v[4] = {1.f, 1.f, 1.f, 1.f};
        for (int itn = 0; itn < 128; ++itn) {
            float nv[4];
            for (int k = 0; k < 4; ++k)
                nv[k] = G[k][0]*v[0] + G[k][1]*v[1] + G[k][2]*v[2] + G[k][3]*v[3];
            float nrm = rsqrtf(nv[0]*nv[0] + nv[1]*nv[1] + nv[2]*nv[2] + nv[3]*nv[3] + 1e-30f);
            for (int k = 0; k < 4; ++k) v[k] = nv[k] * nrm;
        }
        // deterministic sign: largest-magnitude component positive
        int am = 0; float mx = fabsf(v[0]);
        for (int k = 1; k < 4; ++k) if (fabsf(v[k]) > mx) { mx = fabsf(v[k]); am = k; }
        const float sgn = (v[am] < 0.f) ? -1.f : 1.f;

        float* fz = ws + PART_FLOATS + b * 8;
        float bias = 0.f;
        for (int k = 0; k < 4; ++k) {
            float w = sgn * v[k] * isg[k];
            fz[k] = w;
            bias += m[k] * w;
        }
        fz[4] = bias;
    }
}

// ---------------------------------------------------------------------------
// Pass 2: out[r] = dot(a_r, w) - bias. Fully coalesced b128 loads.
// ---------------------------------------------------------------------------
extern "C" __global__ __launch_bounds__(256)
void bpca_pass2(const float4* __restrict__ in4, const float* __restrict__ ws,
                float* __restrict__ out) {
    const int b = blockIdx.y;
    const float* fz = ws + PART_FLOATS + b * 8;
    const float w0 = fz[0], w1 = fz[1], w2 = fz[2], w3 = fz[3], bias = fz[4];
    const int idx = b * MROWS + blockIdx.x * 256 + threadIdx.x;
    const float4 a = in4[idx];
    out[idx] = a.x * w0 + a.y * w1 + a.z * w2 + a.w * w3 - bias;
}

// ---------------------------------------------------------------------------
extern "C" void kernel_launch(void* const* d_in, const int* in_sizes, int n_in,
                              void* d_out, int out_size, void* d_ws, size_t ws_size,
                              hipStream_t stream) {
    const float4* x4 = (const float4*)d_in[0];   // [8,224,224,64] f32, rows of 4 contiguous
    float* ws  = (float*)d_ws;
    float* out = (float*)d_out;

    dim3 g1(NWG, NBATCH);
    hipLaunchKernelGGL(bpca_pass1, g1, dim3(256), 0, stream, x4, ws);
    hipLaunchKernelGGL(bpca_finalize, dim3(NBATCH), dim3(32), 0, stream, ws);
    dim3 g2(MROWS / 256, NBATCH);
    hipLaunchKernelGGL(bpca_pass2, g2, dim3(256), 0, stream, x4, ws, out);
}
